// CFAConv_64948495450338
// MI455X (gfx1250) — matrix operand order repaired and checked by reading, not compiled, not run on hardware
//
#include <hip/hip_runtime.h>
#include <hip/hip_fp16.h>

// CFAConv for MI455X (gfx1250): full f16-WMMA pipeline, flash-fused softmax.
// B=4, C=256, Ch=128, H=W=64, N=4096.
// Round 3: + global_prefetch of the next streaming tile (overlaps L2 refill
// with WMMA issue). Fragments remain pure ds_load_b128 / global_load_b128.

typedef __attribute__((ext_vector_type(16))) _Float16 v16h;
typedef __attribute__((ext_vector_type(8)))  _Float16 v8h;
typedef __attribute__((ext_vector_type(8)))  float    v8f;

#define EPS_BN 1e-5f

// ---- WMMA fragment helpers (layouts per CDNA5 ISA 7.12.2) -----------------
// 16-bit A 16x32: lane m = lane&15, half = lane>>4; element e -> K:
//   e in [0,8)  -> k0 + half*8 + e
//   e in [8,16) -> k0 + 16 + half*8 + (e-8)
// i.e. two contiguous 8-half runs.
__device__ __forceinline__ v16h load_A(const _Float16* base, int stride, int m,
                                       int half, int k0) {
  const _Float16* p = base + m * stride + k0 + (half << 3);
  v8h lo = *(const v8h*)p;
  v8h hi = *(const v8h*)(p + 16);
  return __builtin_shufflevector(lo, hi, 0, 1, 2, 3, 4, 5, 6, 7, 8, 9, 10, 11,
                                 12, 13, 14, 15);
}

// B fragment from [n][K] buffer: same K order as load_A,
// two contiguous 8-half runs at k0 + half*8 and k0 + 16 + half*8.
__device__ __forceinline__ v16h load_B_nk(const _Float16* base, int stride,
                                          int n, int half, int k0) {
  const _Float16* p = base + n * stride + k0 + (half << 3);
  v8h lo = *(const v8h*)p;
  v8h hi = *(const v8h*)(p + 16);
  return __builtin_shufflevector(lo, hi, 0, 1, 2, 3, 4, 5, 6, 7, 8, 9, 10, 11,
                                 12, 13, 14, 15);
}

__device__ __forceinline__ v8f wmma16(v16h a, v16h b, v8f c) {
  return __builtin_amdgcn_wmma_f32_16x16x32_f16(false, a, false, b, (short)0,
                                                c, false, false);
}

__device__ __forceinline__ float gelu_erf(float x) {
  return 0.5f * x * (1.0f + erff(x * 0.7071067811865475f));
}

__device__ __forceinline__ void atomicMaxF(float* addr, float val) {
  unsigned int* ai = (unsigned int*)addr;
  unsigned int old = *ai;
  for (;;) {
    float cur = __uint_as_float(old);
    if (cur >= val) break;
    unsigned int assumed = old;
    old = atomicCAS(ai, assumed, __float_as_uint(val));
    if (old == assumed) break;
  }
}

// ---- prep kernels ----------------------------------------------------------

__global__ void k_init_out(float* out, int n) {
  int i = blockIdx.x * blockDim.x + threadIdx.x;
  if (i < n) out[i] = -INFINITY;
}

__global__ void k_cvt_f16(const float* __restrict__ src,
                          _Float16* __restrict__ dst, int n) {
  int i = blockIdx.x * blockDim.x + threadIdx.x;
  if (i < n) dst[i] = (_Float16)src[i];
}

// [O][IC][3][3] f32 -> [9][O][IC] f16
__global__ void k_wreorder(const float* __restrict__ w,
                           _Float16* __restrict__ wt, int O, int IC) {
  int i = blockIdx.x * blockDim.x + threadIdx.x;
  int total = O * IC * 9;
  if (i >= total) return;
  int tap = i % 9;
  int ci  = (i / 9) % IC;
  int o   = i / (9 * IC);
  wt[((size_t)tap * O + o) * IC + ci] = (_Float16)w[i];
}

// inputs_0 -> channels [256..511] of xcat (f16, NCHW flat)
__global__ void k_fillcat(const float* __restrict__ x,
                          _Float16* __restrict__ xcat) {
  int i = blockIdx.x * blockDim.x + threadIdx.x;
  if (i >= 4 * 256 * 4096) return;
  int p = i & 4095;
  int c = (i >> 12) & 255;
  int b = i >> 20;
  xcat[((size_t)b * 512 + 256 + c) * 4096 + p] = (_Float16)x[i];
}

// ---- projections: x1=[Ch][N], x2t/x3t=[N][Ch] (f16) ------------------------
// grid (N/16, B, 3), block 256

__global__ __launch_bounds__(256) void k_proj(
    const float* __restrict__ in0, const float* __restrict__ in1,
    const _Float16* __restrict__ wq, const _Float16* __restrict__ wk,
    const _Float16* __restrict__ wv, const float* __restrict__ bq,
    const float* __restrict__ bk, const float* __restrict__ bv,
    _Float16* __restrict__ x1, _Float16* __restrict__ x2t,
    _Float16* __restrict__ x3t) {
  __shared__ alignas(32) _Float16 sW[128 * 32];  // [c][ci]
  __shared__ alignas(32) _Float16 sX[16 * 32];   // [p][ci]
  const int tid = threadIdx.x;
  const int lane = tid & 31, l15 = lane & 15, half = lane >> 4, wid = tid >> 5;
  const int pbase = blockIdx.x * 16;
  const int b = blockIdx.y;
  const int proj = blockIdx.z;
  const float* in = (proj == 2) ? in1 : in0;
  const _Float16* w = (proj == 0) ? wq : (proj == 1) ? wk : wv;
  const float* bias = (proj == 0) ? bq : (proj == 1) ? bk : bv;

  v8f acc = {};
  for (int kc = 0; kc < 8; ++kc) {
    const int ci0 = kc * 32;
    // prefetch next chunk of the input image (speculative, L2-bound)
    if (kc < 7)
      __builtin_prefetch(
          in + ((size_t)b * 256 + ci0 + 32 + (tid & 31)) * 4096 + pbase, 0, 3);
    // weights: 32 contiguous halves per row -> v8h moves
    for (int i = tid; i < 128 * 4; i += 256)
      ((v8h*)sW)[i] = *(const v8h*)(w + (i >> 2) * 256 + ci0 + (i & 3) * 8);
    // input tile transposed to [p][ci] so B fragments are contiguous
    for (int i = tid; i < 16 * 32; i += 256)
      sX[(i & 15) * 32 + (i >> 4)] =
          (_Float16)in[((size_t)b * 256 + ci0 + (i >> 4)) * 4096 + pbase + (i & 15)];
    __syncthreads();
    v16h a = load_A(sW + wid * 16 * 32, 32, l15, half, 0);
    v16h bb = load_B_nk(sX, 32, l15, half, 0);
    acc = wmma16(a, bb, acc);
    __syncthreads();
  }
  const int p = pbase + l15;
  if (proj == 0) {
#pragma unroll
    for (int r = 0; r < 8; ++r) {
      int c = wid * 16 + r + (half << 3);
      x1[((size_t)b * 128 + c) * 4096 + p] = (_Float16)(acc[r] + bias[c]);
    }
  } else {
    v8h ov;
#pragma unroll
    for (int r = 0; r < 8; ++r)
      ov[r] = (_Float16)(acc[r] + bias[wid * 16 + r + (half << 3)]);
    _Float16* dst = (proj == 1) ? x2t : x3t;
    *(v8h*)(dst + ((size_t)b * 4096 + p) * 128 + wid * 16 + (half << 3)) = ov;
  }
}

// ---- flash-fused attention -------------------------------------------------
// aff[j,k] = sum_i x2[i,j]*x3[i,k]; softmax over j; x0[c,k]=sum_j x1[c,j]P[j,k]
// grid (N/32, B), block 256. Block owns 32 columns k; streams j in tiles of 64.

__global__ __launch_bounds__(256) void k_attn(
    const _Float16* __restrict__ x1, const _Float16* __restrict__ x2t,
    const _Float16* __restrict__ x3t, _Float16* __restrict__ x0t) {
  __shared__ alignas(32) _Float16 sQ[32 * 128];  // x3t tile [k][c]
  __shared__ alignas(32) _Float16 sK[64 * 128];  // x2t tile [j][c]
  __shared__ alignas(32) _Float16 sV[128 * 64];  // x1 tile  [c][j]
  __shared__ alignas(32) _Float16 sP[32 * 64];   // probs    [k][j]
  __shared__ float sRed[4 * 32], sSum[4 * 32];
  __shared__ float sM[32], sL[32], sAlpha[32], sMnew[32];

  const int tid = threadIdx.x;
  const int lane = tid & 31, l15 = lane & 15, half = lane >> 4, wid = tid >> 5;
  const int b = blockIdx.y;
  const int colbase = blockIdx.x * 32;
  const int sm = wid & 3;   // j-tile of this wave's S tile
  const int sn = wid >> 2;  // column tile (0/1)
  const int col = sn * 16 + l15;
  const int mt0 = (wid & 3) * 2, mt1 = mt0 + 1;  // this wave's O c-tiles

  {  // x3t tile is one contiguous 4096-half block
    const _Float16* src = x3t + ((size_t)b * 4096 + colbase) * 128;
    for (int i = tid; i < 32 * 128 / 8; i += 256)
      ((v8h*)sQ)[i] = *(const v8h*)(src + i * 8);
  }
  if (tid < 32) {
    sM[tid] = -INFINITY;
    sL[tid] = 0.0f;
  }
  __syncthreads();

  v8f o0 = {}, o1 = {};

  for (int jb = 0; jb < 4096; jb += 64) {
    {  // x2t tile contiguous
      const _Float16* src = x2t + ((size_t)b * 4096 + jb) * 128;
      for (int i = tid; i < 64 * 128 / 8; i += 256)
        ((v8h*)sK)[i] = *(const v8h*)(src + i * 8);
    }
    // x1 tile: 128 rows of 64 contiguous halves
    for (int i = tid; i < 128 * 64 / 8; i += 256)
      ((v8h*)sV)[i] = *(const v8h*)(x1 + ((size_t)b * 128 + (i >> 3)) * 4096 +
                                    jb + (i & 7) * 8);
    // prefetch next j-tile while this one computes (L2-resident stream)
    if (jb + 64 < 4096) {
      __builtin_prefetch(x2t + ((size_t)b * 4096 + jb + 64) * 128 + tid * 32,
                         0, 3);
      __builtin_prefetch(
          x1 + ((size_t)b * 128 + (tid & 127)) * 4096 + jb + 64, 0, 3);
    }
    __syncthreads();

    // S = x2^T x3 over the 128-ch contraction (4 WMMA k-steps)
    v8f s = {};
#pragma unroll
    for (int kc = 0; kc < 4; ++kc) {
      v16h a = load_A(sK + sm * 16 * 128, 128, l15, half, kc * 32);
      v16h bb = load_B_nk(sQ + sn * 16 * 128, 128, l15, half, kc * 32);
      s = wmma16(a, bb, s);
    }

    // online-softmax column max
    float mx = s[0];
#pragma unroll
    for (int r = 1; r < 8; ++r) mx = fmaxf(mx, s[r]);
    mx = fmaxf(mx, __shfl_xor(mx, 16, 32));
    if (lane < 16) sRed[sm * 32 + col] = mx;
    __syncthreads();

    if (tid < 32) {
      float cand = fmaxf(fmaxf(sRed[tid], sRed[32 + tid]),
                         fmaxf(sRed[64 + tid], sRed[96 + tid]));
      float mold = sM[tid];
      float mnew = fmaxf(mold, cand);
      sMnew[tid] = mnew;
      sAlpha[tid] = __expf(mold - mnew);
      sM[tid] = mnew;
    }
    __syncthreads();

    // P = exp(S - mnew); rescale O by alpha; row-sum for l.
    // sP is [k][j]: D-layout r walks j contiguously -> one v8h store.
    const float mn = sMnew[col];
    const float al = sAlpha[col];
    float psum = 0.0f;
    v8h pvec;
#pragma unroll
    for (int r = 0; r < 8; ++r) {
      float pv = __expf(s[r] - mn);
      psum += pv;
      pvec[r] = (_Float16)pv;
      o0[r] *= al;
      o1[r] *= al;
    }
    *(v8h*)(sP + col * 64 + sm * 16 + (half << 3)) = pvec;
    psum += __shfl_xor(psum, 16, 32);
    if (lane < 16) sSum[sm * 32 + col] = psum;
    __syncthreads();

    if (tid < 32)
      sL[tid] = sL[tid] * sAlpha[tid] + sSum[tid] + sSum[32 + tid] +
                sSum[64 + tid] + sSum[96 + tid];

    // O += x1 * P (K = 64 j -> 2 WMMA k-steps); B from [k][j] is contiguous
#pragma unroll
    for (int kc = 0; kc < 2; ++kc) {
      v16h bb = load_B_nk(sP, 64, col, half, kc * 32);
      v16h a0 = load_A(sV + mt0 * 16 * 64, 64, l15, half, kc * 32);
      o0 = wmma16(a0, bb, o0);
      v16h a1 = load_A(sV + mt1 * 16 * 64, 64, l15, half, kc * 32);
      o1 = wmma16(a1, bb, o1);
    }
    __syncthreads();
  }

  const float invl = 1.0f / sL[col];
  _Float16* dst = x0t + ((size_t)b * 4096 + colbase + col) * 128;
  v8h w0, w1;
#pragma unroll
  for (int r = 0; r < 8; ++r) {
    w0[r] = (_Float16)(o0[r] * invl);
    w1[r] = (_Float16)(o1[r] * invl);
  }
  *(v8h*)(dst + mt0 * 16 + (half << 3)) = w0;
  *(v8h*)(dst + mt1 * 16 + (half << 3)) = w1;
}

// ---- output projection 128->256 + BN + GELU --------------------------------
// grid (N/16, B), block 256. Writes x0g (f32) and xcat channels [0..255] (f16)

__global__ __launch_bounds__(256) void k_oproj(
    const _Float16* __restrict__ x0t, const _Float16* __restrict__ wo,
    const float* __restrict__ bo, const float* __restrict__ g,
    const float* __restrict__ bb, const float* __restrict__ m,
    const float* __restrict__ vv, float* __restrict__ x0g,
    _Float16* __restrict__ xcat) {
  __shared__ alignas(32) _Float16 sW[256 * 32];
  const int tid = threadIdx.x;
  const int lane = tid & 31, l15 = lane & 15, half = lane >> 4, wid = tid >> 5;
  const int pbase = blockIdx.x * 16;
  const int b = blockIdx.y;
  const int mt0 = wid * 2, mt1 = mt0 + 1;

  v8f acc0 = {}, acc1 = {};
  for (int kc = 0; kc < 4; ++kc) {
    for (int i = tid; i < 256 * 4; i += 256)
      ((v8h*)sW)[i] = *(const v8h*)(wo + (i >> 2) * 128 + kc * 32 + (i & 3) * 8);
    __syncthreads();
    // contiguous B fragment straight from global [p][c] layout
    v16h bf = load_B_nk(x0t + ((size_t)(b * 4096 + pbase)) * 128, 128, l15,
                        half, kc * 32);
    v16h a0 = load_A(sW + mt0 * 16 * 32, 32, l15, half, 0);
    acc0 = wmma16(a0, bf, acc0);
    v16h a1 = load_A(sW + mt1 * 16 * 32, 32, l15, half, 0);
    acc1 = wmma16(a1, bf, acc1);
    __syncthreads();
  }
  const int p = pbase + l15;
#pragma unroll
  for (int r = 0; r < 8; ++r) {
#pragma unroll
    for (int t = 0; t < 2; ++t) {
      int o = (t ? mt1 : mt0) * 16 + r + (half << 3);
      float val = (t ? acc1[r] : acc0[r]) + bo[o];
      float sc = g[o] * rsqrtf(vv[o] + EPS_BN);
      val = (val - m[o]) * sc + bb[o];
      val = gelu_erf(val);
      x0g[((size_t)b * 256 + o) * 4096 + p] = val;
      xcat[((size_t)b * 512 + o) * 4096 + p] = (_Float16)val;
    }
  }
}

// ---- 3x3 conv as implicit GEMM + BN + GELU (+fused residual & max) ---------
// grid (W/16, H, B*2), block 256. Block: 128 out-ch x 16 pixels of one row.

template <int IC, bool FINAL>
__global__ __launch_bounds__(256) void k_conv3(
    const _Float16* __restrict__ src, const _Float16* __restrict__ wt,
    const float* __restrict__ bias, const float* __restrict__ g,
    const float* __restrict__ bb, const float* __restrict__ m,
    const float* __restrict__ vv, _Float16* __restrict__ dst,
    const float* __restrict__ x0g, float* __restrict__ out) {
  // halo tile [3 rows][18 cols][32 ci]: B fragments contiguous in ci
  __shared__ alignas(32) _Float16 sIn[3 * 18 * 32];
  __shared__ alignas(32) _Float16 sWt[128 * 32];
  const int tid = threadIdx.x;
  const int lane = tid & 31, l15 = lane & 15, half = lane >> 4, wid = tid >> 5;
  const int px0 = blockIdx.x * 16;
  const int y = blockIdx.y;
  const int b = blockIdx.z >> 1;
  const int og = blockIdx.z & 1;

  v8f acc = {};
  for (int cc = 0; cc < IC / 32; ++cc) {
    // prefetch next chunk's center row (speculative)
    if (cc + 1 < IC / 32)
      __builtin_prefetch(
          src + ((size_t)b * IC + (cc + 1) * 32 + (tid & 31)) * 4096 + y * 64 +
              px0,
          0, 3);
    for (int i = tid; i < 54 * 32; i += 256) {
      int ci = i & 31, pos = i >> 5, row = pos / 18, colt = pos % 18;
      int iy = y + row - 1, ix = px0 + colt - 1;
      _Float16 vval = (_Float16)0.0f;
      if (iy >= 0 && iy < 64 && ix >= 0 && ix < 64)
        vval = src[((size_t)b * IC + cc * 32 + ci) * 4096 + iy * 64 + ix];
      sIn[pos * 32 + ci] = vval;
    }
    __syncthreads();
    for (int tap = 0; tap < 9; ++tap) {
      const int kh = tap / 3, kw = tap % 3;
      for (int i = tid; i < 128 * 4; i += 256)
        ((v8h*)sWt)[i] =
            *(const v8h*)(wt + ((size_t)(tap * 256 + og * 128 + (i >> 2))) * IC +
                          cc * 32 + (i & 3) * 8);
      __syncthreads();
      v16h a = load_A(sWt + wid * 16 * 32, 32, l15, half, 0);
      v16h bf = load_B_nk(sIn, 32, kh * 18 + l15 + kw, half, 0);
      acc = wmma16(a, bf, acc);
      __syncthreads();
    }
  }
  const int p = y * 64 + px0 + l15;
  float vals[8];
#pragma unroll
  for (int r = 0; r < 8; ++r) {
    int o = og * 128 + wid * 16 + r + (half << 3);
    float val = acc[r] + bias[o];
    float sc = g[o] * rsqrtf(vv[o] + EPS_BN);
    val = (val - m[o]) * sc + bb[o];
    val = gelu_erf(val);
    if (FINAL) {
      vals[r] = val + x0g[((size_t)b * 256 + o) * 4096 + p];
    } else {
      dst[((size_t)b * 256 + o) * 4096 + p] = (_Float16)val;
    }
  }
  if (FINAL) {
#pragma unroll
    for (int r = 0; r < 8; ++r) {
      float mx = vals[r];
#pragma unroll
      for (int off = 8; off >= 1; off >>= 1)
        mx = fmaxf(mx, __shfl_xor(mx, off, 32));  // stays within 16-lane half
      if (l15 == 0) {
        int o = og * 128 + wid * 16 + r + (half << 3);
        atomicMaxF(&out[b * 256 + o], mx);
      }
    }
  }
}

// ---- host launch -----------------------------------------------------------

extern "C" void kernel_launch(void* const* d_in, const int* in_sizes, int n_in,
                              void* d_out, int out_size, void* d_ws,
                              size_t ws_size, hipStream_t stream) {
  const float* in0 = (const float*)d_in[0];
  const float* in1 = (const float*)d_in[1];
  const float* w_q = (const float*)d_in[2];
  const float* b_q = (const float*)d_in[3];
  const float* w_k = (const float*)d_in[4];
  const float* b_k = (const float*)d_in[5];
  const float* w_v = (const float*)d_in[6];
  const float* b_v = (const float*)d_in[7];
  const float* w_o = (const float*)d_in[8];
  const float* b_o = (const float*)d_in[9];
  const float* bn0_g = (const float*)d_in[10];
  const float* bn0_b = (const float*)d_in[11];
  const float* bn0_m = (const float*)d_in[12];
  const float* bn0_v = (const float*)d_in[13];
  const float* cb_w0 = (const float*)d_in[14];
  const float* cb_b0 = (const float*)d_in[15];
  const float* cb_bn0_g = (const float*)d_in[16];
  const float* cb_bn0_b = (const float*)d_in[17];
  const float* cb_bn0_m = (const float*)d_in[18];
  const float* cb_bn0_v = (const float*)d_in[19];
  const float* cb_w1 = (const float*)d_in[20];
  const float* cb_b1 = (const float*)d_in[21];
  const float* cb_bn1_g = (const float*)d_in[22];
  const float* cb_bn1_b = (const float*)d_in[23];
  const float* cb_bn1_m = (const float*)d_in[24];
  const float* cb_bn1_v = (const float*)d_in[25];
  float* out = (float*)d_out;

  char* ws = (char*)d_ws;
  size_t off = 0;
  auto take = [&](size_t bytes) -> char* {
    char* p = ws + off;
    off += (bytes + 255) & ~(size_t)255;
    return p;
  };
  _Float16* wq16 = (_Float16*)take((size_t)128 * 256 * 2);
  _Float16* wk16 = (_Float16*)take((size_t)128 * 256 * 2);
  _Float16* wv16 = (_Float16*)take((size_t)128 * 256 * 2);
  _Float16* wo16 = (_Float16*)take((size_t)256 * 128 * 2);
  _Float16* wt0 = (_Float16*)take((size_t)9 * 256 * 512 * 2);
  _Float16* wt1 = (_Float16*)take((size_t)9 * 256 * 256 * 2);
  _Float16* x1 = (_Float16*)take((size_t)4 * 128 * 4096 * 2);
  _Float16* x2t = (_Float16*)take((size_t)4 * 4096 * 128 * 2);
  _Float16* x3t = (_Float16*)take((size_t)4 * 4096 * 128 * 2);
  _Float16* x0t = (_Float16*)take((size_t)4 * 4096 * 128 * 2);
  float* x0g = (float*)take((size_t)4 * 256 * 4096 * 4);
  _Float16* xcat = (_Float16*)take((size_t)4 * 512 * 4096 * 2);
  _Float16* y1 = (_Float16*)take((size_t)4 * 256 * 4096 * 2);
  (void)ws_size;
  (void)n_in;
  (void)in_sizes;

  k_init_out<<<(out_size + 255) / 256, 256, 0, stream>>>(out, out_size);
  k_cvt_f16<<<(32768 + 255) / 256, 256, 0, stream>>>(w_q, wq16, 32768);
  k_cvt_f16<<<(32768 + 255) / 256, 256, 0, stream>>>(w_k, wk16, 32768);
  k_cvt_f16<<<(32768 + 255) / 256, 256, 0, stream>>>(w_v, wv16, 32768);
  k_cvt_f16<<<(32768 + 255) / 256, 256, 0, stream>>>(w_o, wo16, 32768);
  k_wreorder<<<(256 * 512 * 9 + 255) / 256, 256, 0, stream>>>(cb_w0, wt0, 256,
                                                              512);
  k_wreorder<<<(256 * 256 * 9 + 255) / 256, 256, 0, stream>>>(cb_w1, wt1, 256,
                                                              256);
  k_fillcat<<<(4 * 256 * 4096 + 255) / 256, 256, 0, stream>>>(in0, xcat);

  k_proj<<<dim3(256, 4, 3), 256, 0, stream>>>(in0, in1, wq16, wk16, wv16, b_q,
                                              b_k, b_v, x1, x2t, x3t);
  k_attn<<<dim3(128, 4), 256, 0, stream>>>(x1, x2t, x3t, x0t);
  k_oproj<<<dim3(256, 4), 256, 0, stream>>>(x0t, wo16, b_o, bn0_g, bn0_b,
                                            bn0_m, bn0_v, x0g, xcat);
  k_conv3<512, false><<<dim3(4, 64, 8), 256, 0, stream>>>(
      xcat, wt0, cb_b0, cb_bn0_g, cb_bn0_b, cb_bn0_m, cb_bn0_v, y1, nullptr,
      nullptr);
  k_conv3<256, true><<<dim3(4, 64, 8), 256, 0, stream>>>(
      y1, wt1, cb_b1, cb_bn1_g, cb_bn1_b, cb_bn1_m, cb_bn1_v, nullptr, x0g,
      out);
}
